// MemoryReader_50397146251683
// MI455X (gfx1250) — compile-verified, matrix operand order checked by reading, not run on hardware
//
#include <hip/hip_runtime.h>
#include <math.h>

typedef __attribute__((ext_vector_type(16))) __bf16  v16bf;
typedef __attribute__((ext_vector_type(8)))  float   v8f;
typedef __attribute__((ext_vector_type(4)))  unsigned int u4;
typedef __attribute__((ext_vector_type(2)))  unsigned int u2;
typedef __attribute__((ext_vector_type(4)))  float   f4;

union Frag {
    v16bf v;
    u4 q[2];
    unsigned int w[8];
    unsigned short h[16];
};

// pack two f32 into bf16x2 (truncate) with a single v_perm_b32
__device__ __forceinline__ unsigned int pkbf(float lo, float hi_) {
    return __builtin_amdgcn_perm(__float_as_uint(hi_), __float_as_uint(lo), 0x07060302u);
}

// ===========================================================================
// FAST PATH (workspace-resident bf16 operands), 64-column tiles, 16 waves.
//   mkt: [b][m][64] bf16 (pre-transposed)   qkt: [b][n][64] bf16
//   mvb: [b][c][m]  bf16                     asq: [b][m] f32
// Per (batch, 64-col tile): stream m in 64-row steps; online softmax.
// Wave w: computes score tile (mt=w>>2, nt=w&3) AND owns CV rows
// [w*32, w*32+32) for all 64 columns -> each MV fragment feeds 4 WMMAs.
// ===========================================================================
__global__ __launch_bounds__(512)
void memreader_flash_fast(const unsigned short* __restrict__ mkt,
                          const unsigned short* __restrict__ qkt,
                          const unsigned short* __restrict__ mvb,
                          const float* __restrict__ asq,
                          float* __restrict__ out)
{
    constexpr int CK = 64, CV = 512, M = 4096, N = 4096, OC = 1024;

    __shared__ __align__(16) unsigned short s_p[64][72];   // [n][m] bf16
    __shared__ float s_cmax[4][64];
    __shared__ float s_psum[64], s_scale[64], s_nmax[64], s_rm[64], s_ls[64];

    const int tid  = threadIdx.x;
    const int lane = tid & 31;
    const int wave = tid >> 5;           // 0..15
    const int l16  = lane & 15;
    const int hi   = lane >> 4;
    const int b    = blockIdx.z;
    const int n0   = blockIdx.x * 64;
    const int mt   = wave >> 2;          // score tile row (0..3)
    const int nt   = wave & 3;           // score tile col (0..3)

    v8f acc[2][4];                       // [rt][ct], rows wave*32+rt*16
    #pragma unroll
    for (int rt = 0; rt < 2; ++rt)
        #pragma unroll
        for (int ct = 0; ct < 4; ++ct)
            #pragma unroll
            for (int r = 0; r < 8; ++r) acc[rt][ct][r] = 0.0f;

    // ---- loop-invariant QK B-fragments (registers for all 64 steps) ----
    Frag qb0, qb1;
    {
        const unsigned short* br = qkt + ((size_t)(b * N + n0 + nt * 16 + l16)) * CK;
        qb0.q[0] = *(const u4*)(br + hi * 8);
        qb0.q[1] = *(const u4*)(br + 16 + hi * 8);
        qb1.q[0] = *(const u4*)(br + 32 + hi * 8);
        qb1.q[1] = *(const u4*)(br + 48 + hi * 8);
    }
    if (tid < 64) { s_rm[tid] = -3.0e38f; s_ls[tid] = 0.0f; }

    // hot-loop pointers (incremented, not recomputed)
    const unsigned short* ar  = mkt + ((size_t)(b * M + mt * 16 + l16)) * CK;  // += 64*CK
    const float*          aqp = asq + (size_t)b * M + mt * 16 + hi * 8;        // += 64
    const unsigned short* mv0 = mvb + ((size_t)(b * CV + wave * 32 + l16)) * M;        // += 64
    const unsigned short* mv1 = mvb + ((size_t)(b * CV + wave * 32 + 16 + l16)) * M;   // += 64

    float t[8];

    for (int it = 0; it < 64; ++it) {
        // ---- scores: S(16x16) = MK(16x64) * QK(64x16), A direct from global ----
        {
            Frag a0, a1;
            a0.q[0] = *(const u4*)(ar + hi * 8);
            a0.q[1] = *(const u4*)(ar + 16 + hi * 8);
            a1.q[0] = *(const u4*)(ar + 32 + hi * 8);
            a1.q[1] = *(const u4*)(ar + 48 + hi * 8);
            if (it + 1 < 64) __builtin_prefetch(ar + 64 * CK, 0, 1);

            v8f s;
            #pragma unroll
            for (int r = 0; r < 8; ++r) s[r] = 0.0f;
            s = __builtin_amdgcn_wmma_f32_16x16x32_bf16(false, a0.v, false, qb0.v, (short)0, s, false, false);
            s = __builtin_amdgcn_wmma_f32_16x16x32_bf16(false, a1.v, false, qb1.v, (short)0, s, false, false);

            f4 aq0 = *(const f4*)(aqp);
            f4 aq1 = *(const f4*)(aqp + 4);
            float aqv[8] = {aq0.x, aq0.y, aq0.z, aq0.w, aq1.x, aq1.y, aq1.z, aq1.w};

            float cmax = -3.0e38f;
            #pragma unroll
            for (int r = 0; r < 8; ++r) {
                float val = (2.0f * s[r] - aqv[r]) * 0.125f;   // 1/sqrt(CK)=1/8
                t[r] = val;
                cmax = fmaxf(cmax, val);
            }
            cmax = fmaxf(cmax, __shfl_xor(cmax, 16, 32));
            if (hi == 0) s_cmax[mt][nt * 16 + l16] = cmax;
        }
        __syncthreads();                                   // (A)

        if (tid < 64) {                                    // online softmax state
            float tm = fmaxf(fmaxf(s_cmax[0][tid], s_cmax[1][tid]),
                             fmaxf(s_cmax[2][tid], s_cmax[3][tid]));
            float rm = s_rm[tid];
            float nm = fmaxf(rm, tm);
            s_scale[tid] = __expf(rm - nm);
            s_nmax[tid]  = nm;
            s_rm[tid]    = nm;
            s_psum[tid]  = 0.0f;
        }
        __syncthreads();                                   // (B)

        // ---- rescale accumulators; P = exp(S-max) -> LDS (B-frag layout) ----
        {
            #pragma unroll
            for (int ct = 0; ct < 4; ++ct) {
                float sc = s_scale[ct * 16 + l16];
                #pragma unroll
                for (int rt = 0; rt < 2; ++rt)
                    #pragma unroll
                    for (int r = 0; r < 8; ++r) acc[rt][ct][r] *= sc;
            }
            float nm = s_nmax[nt * 16 + l16];
            float pe[8];
            float psum = 0.0f;
            #pragma unroll
            for (int r = 0; r < 8; ++r) {
                pe[r] = __expf(t[r] - nm);
                psum += pe[r];
            }
            Frag p;
            #pragma unroll
            for (int j = 0; j < 4; ++j) p.w[j] = pkbf(pe[2 * j], pe[2 * j + 1]);
            *(u4*)(&s_p[nt * 16 + l16][mt * 16 + hi * 8]) = p.q[0];
            atomicAdd(&s_psum[nt * 16 + l16], psum);
        }
        __syncthreads();                                   // (C)
        if (tid < 64) s_ls[tid] = s_ls[tid] * s_scale[tid] + s_psum[tid];

        // ---- O(32x64) += MV(32x64) * P(64x64): 16 WMMA per wave ----
        {
            Frag a[2][2];                                  // [rt][ks]
            #pragma unroll
            for (int ks = 0; ks < 2; ++ks) {
                a[0][ks].q[0] = *(const u4*)(mv0 + ks * 32 + hi * 8);
                a[0][ks].q[1] = *(const u4*)(mv0 + ks * 32 + 16 + hi * 8);
                a[1][ks].q[0] = *(const u4*)(mv1 + ks * 32 + hi * 8);
                a[1][ks].q[1] = *(const u4*)(mv1 + ks * 32 + 16 + hi * 8);
            }
            #pragma unroll
            for (int ct = 0; ct < 4; ++ct) {
                const unsigned short* pr = s_p[ct * 16 + l16];
                Frag f0, f1;
                f0.q[0] = *(const u4*)(pr + hi * 8);
                f0.q[1] = *(const u4*)(pr + 16 + hi * 8);
                f1.q[0] = *(const u4*)(pr + 32 + hi * 8);
                f1.q[1] = *(const u4*)(pr + 48 + hi * 8);
                #pragma unroll
                for (int rt = 0; rt < 2; ++rt) {
                    acc[rt][ct] = __builtin_amdgcn_wmma_f32_16x16x32_bf16(
                        false, a[rt][0].v, false, f0.v, (short)0, acc[rt][ct], false, false);
                    acc[rt][ct] = __builtin_amdgcn_wmma_f32_16x16x32_bf16(
                        false, a[rt][1].v, false, f1.v, (short)0, acc[rt][ct], false, false);
                }
            }
        }
        ar  += 64 * CK;
        aqp += 64;
        mv0 += 64;
        mv1 += 64;
    }
    __syncthreads();

    // ---- epilogue: normalize and store ----
    #pragma unroll
    for (int ct = 0; ct < 4; ++ct) {
        float inv = 1.0f / s_ls[ct * 16 + l16];
        int n = n0 + ct * 16 + l16;
        #pragma unroll
        for (int rt = 0; rt < 2; ++rt) {
            int cbase = wave * 32 + rt * 16 + hi * 8;
            #pragma unroll
            for (int r = 0; r < 8; ++r)
                out[((size_t)(b * OC + cbase + r)) * N + n] = acc[rt][ct][r] * inv;
        }
    }
}

// ===========================================================================
// Precompute kernels (fast path)
// ===========================================================================

// mv f32 -> bf16, layout preserved
__global__ __launch_bounds__(256)
void kprep_mv_bf16(const f4* __restrict__ in, u2* __restrict__ outp)
{
    size_t i = (size_t)blockIdx.x * blockDim.x + threadIdx.x;  // 4,194,304 total
    f4 x = in[i];
    u2 o;
    o.x = pkbf(x.x, x.y);
    o.y = pkbf(x.z, x.w);
    outp[i] = o;
}

// [b][64][4096] f32 -> [b][4096][64] bf16 (+ optional |row|^2 over c)
__global__ __launch_bounds__(256)
void kprep_transpose(const float* __restrict__ src, unsigned short* __restrict__ dst,
                     float* __restrict__ asqout)
{
    __shared__ __align__(16) unsigned short s_t[64][72];
    __shared__ float s_a[64];
    const int tid = threadIdx.x;
    const int b   = blockIdx.z;
    const int m0  = blockIdx.x * 64;

    if (tid < 64) s_a[tid] = 0.0f;
    __syncthreads();

    const int m = tid & 63;
    float asqp = 0.0f;
    #pragma unroll
    for (int i = 0; i < 8; ++i) {
        int c = ((tid >> 6) << 1) + 8 * i;                 // even c
        float v0 = src[((size_t)(b * 64 + c    )) * 4096 + m0 + m];
        float v1 = src[((size_t)(b * 64 + c + 1)) * 4096 + m0 + m];
        *(unsigned int*)&s_t[m][c] = pkbf(v0, v1);
        asqp += v0 * v0 + v1 * v1;
    }
    if (asqout) atomicAdd(&s_a[m], asqp);
    __syncthreads();

    {   // coalesced writeback: 4 threads per row, 32B each
        int row = tid >> 2;
        int col = (tid & 3) * 16;
        const u4* sp = (const u4*)&s_t[row][col];
        u4* dp = (u4*)(dst + ((size_t)(b * 4096 + m0 + row)) * 64 + col);
        dp[0] = sp[0];
        dp[1] = sp[1];
    }
    if (asqout && tid < 64) asqout[(size_t)b * 4096 + m0 + tid] = s_a[tid];
}

// ===========================================================================
// FALLBACK (no workspace): in-kernel staging, f32 mv with perm conversion
// ===========================================================================
__global__ __launch_bounds__(256)
void memreader_flash_fallback(const float* __restrict__ mk,
                              const float* __restrict__ qk,
                              const float* __restrict__ mv,
                              float* __restrict__ out)
{
    constexpr int CK = 64, CV = 512, M = 4096, N = 4096, OC = 1024;

    __shared__ __align__(16) unsigned short s_qk[32][72];
    __shared__ __align__(16) unsigned short s_mk[64][72];
    __shared__ __align__(16) unsigned short s_p [32][72];
    __shared__ float s_asq[64];
    __shared__ float s_cmax[4][32];
    __shared__ float s_psum[32], s_scale[32], s_nmax[32], s_rm[32], s_ls[32];

    const int tid  = threadIdx.x;
    const int lane = tid & 31;
    const int wave = tid >> 5;
    const int l16  = lane & 15;
    const int hi   = lane >> 4;
    const int b    = blockIdx.z;
    const int n0   = blockIdx.x * 32;
    const int mt   = wave >> 1;
    const int nt   = wave & 1;

    v8f acc[4][2];
    #pragma unroll
    for (int rt = 0; rt < 4; ++rt)
        #pragma unroll
        for (int ct = 0; ct < 2; ++ct)
            #pragma unroll
            for (int r = 0; r < 8; ++r) acc[rt][ct][r] = 0.0f;

    {
        const int nq = tid & 31;
        #pragma unroll
        for (int i = 0; i < 4; ++i) {
            int c = ((tid >> 5) << 1) + 16 * i;
            float v0 = qk[((size_t)(b * CK + c    )) * N + n0 + nq];
            float v1 = qk[((size_t)(b * CK + c + 1)) * N + n0 + nq];
            *(unsigned int*)&s_qk[nq][c] = pkbf(v0, v1);
        }
    }
    if (tid < 32) { s_rm[tid] = -3.0e38f; s_ls[tid] = 0.0f; }

    float t[8];

    for (int it = 0; it < 64; ++it) {
        const int m0 = it * 64;
        __syncthreads();
        if (tid < 64) s_asq[tid] = 0.0f;
        __syncthreads();
        {
            const int m = tid & 63;
            float asqp = 0.0f;
            #pragma unroll
            for (int i = 0; i < 8; ++i) {
                int c = ((tid >> 6) << 1) + 8 * i;
                float v0 = mk[((size_t)(b * CK + c    )) * M + m0 + m];
                float v1 = mk[((size_t)(b * CK + c + 1)) * M + m0 + m];
                *(unsigned int*)&s_mk[m][c] = pkbf(v0, v1);
                asqp += v0 * v0 + v1 * v1;
            }
            atomicAdd(&s_asq[m], asqp);
        }
        __syncthreads();
        {
            Frag a0, a1, b0, b1;
            const unsigned short* ar = s_mk[mt * 16 + l16];
            const unsigned short* br = s_qk[nt * 16 + l16];
            a0.q[0] = *(const u4*)(ar + hi * 8);
            a0.q[1] = *(const u4*)(ar + 16 + hi * 8);
            a1.q[0] = *(const u4*)(ar + 32 + hi * 8);
            a1.q[1] = *(const u4*)(ar + 48 + hi * 8);
            b0.q[0] = *(const u4*)(br + hi * 8);
            b0.q[1] = *(const u4*)(br + 16 + hi * 8);
            b1.q[0] = *(const u4*)(br + 32 + hi * 8);
            b1.q[1] = *(const u4*)(br + 48 + hi * 8);
            v8f s;
            #pragma unroll
            for (int r = 0; r < 8; ++r) s[r] = 0.0f;
            s = __builtin_amdgcn_wmma_f32_16x16x32_bf16(false, a0.v, false, b0.v, (short)0, s, false, false);
            s = __builtin_amdgcn_wmma_f32_16x16x32_bf16(false, a1.v, false, b1.v, (short)0, s, false, false);
            float cmax = -3.0e38f;
            #pragma unroll
            for (int r = 0; r < 8; ++r) {
                float aq  = s_asq[mt * 16 + hi * 8 + r];
                float val = (2.0f * s[r] - aq) * 0.125f;
                t[r] = val;
                cmax = fmaxf(cmax, val);
            }
            cmax = fmaxf(cmax, __shfl_xor(cmax, 16, 32));
            if (hi == 0) s_cmax[mt][nt * 16 + l16] = cmax;
        }
        __syncthreads();
        if (tid < 32) {
            float tm = fmaxf(fmaxf(s_cmax[0][tid], s_cmax[1][tid]),
                             fmaxf(s_cmax[2][tid], s_cmax[3][tid]));
            float rm = s_rm[tid];
            float nm = fmaxf(rm, tm);
            s_scale[tid] = __expf(rm - nm);
            s_nmax[tid]  = nm;
            s_rm[tid]    = nm;
            s_psum[tid]  = 0.0f;
        }
        __syncthreads();
        {
            #pragma unroll
            for (int ct = 0; ct < 2; ++ct) {
                float sc = s_scale[ct * 16 + l16];
                #pragma unroll
                for (int rt = 0; rt < 4; ++rt)
                    #pragma unroll
                    for (int r = 0; r < 8; ++r) acc[rt][ct][r] *= sc;
            }
            float nm = s_nmax[nt * 16 + l16];
            float pe[8];
            float psum = 0.0f;
            #pragma unroll
            for (int r = 0; r < 8; ++r) {
                pe[r] = __expf(t[r] - nm);
                psum += pe[r];
            }
            Frag p;
            #pragma unroll
            for (int j = 0; j < 4; ++j) p.w[j] = pkbf(pe[2 * j], pe[2 * j + 1]);
            *(u4*)(&s_p[nt * 16 + l16][mt * 16 + hi * 8]) = p.q[0];
            atomicAdd(&s_psum[nt * 16 + l16], psum);
        }
        __syncthreads();
        if (tid < 32) s_ls[tid] = s_ls[tid] * s_scale[tid] + s_psum[tid];
        {
            Frag bfr[2][2];
            #pragma unroll
            for (int ct = 0; ct < 2; ++ct) {
                const unsigned short* pr = s_p[ct * 16 + l16];
                #pragma unroll
                for (int ks = 0; ks < 2; ++ks) {
                    bfr[ct][ks].q[0] = *(const u4*)(pr + ks * 32 + hi * 8);
                    bfr[ct][ks].q[1] = *(const u4*)(pr + ks * 32 + 16 + hi * 8);
                }
            }
            #pragma unroll
            for (int rt = 0; rt < 4; ++rt) {
                const float* mrow = mv + ((size_t)(b * CV + wave * 64 + rt * 16 + l16)) * M + m0;
                Frag a[2];
                #pragma unroll
                for (int ks = 0; ks < 2; ++ks) {
                    f4 x0 = *(const f4*)(mrow + ks * 32 + hi * 8);
                    f4 x1 = *(const f4*)(mrow + ks * 32 + hi * 8 + 4);
                    f4 y0 = *(const f4*)(mrow + ks * 32 + 16 + hi * 8);
                    f4 y1 = *(const f4*)(mrow + ks * 32 + 16 + hi * 8 + 4);
                    a[ks].w[0] = pkbf(x0.x, x0.y); a[ks].w[1] = pkbf(x0.z, x0.w);
                    a[ks].w[2] = pkbf(x1.x, x1.y); a[ks].w[3] = pkbf(x1.z, x1.w);
                    a[ks].w[4] = pkbf(y0.x, y0.y); a[ks].w[5] = pkbf(y0.z, y0.w);
                    a[ks].w[6] = pkbf(y1.x, y1.y); a[ks].w[7] = pkbf(y1.z, y1.w);
                }
                #pragma unroll
                for (int ct = 0; ct < 2; ++ct)
                    #pragma unroll
                    for (int ks = 0; ks < 2; ++ks)
                        acc[rt][ct] = __builtin_amdgcn_wmma_f32_16x16x32_bf16(
                            false, a[ks].v, false, bfr[ct][ks].v, (short)0,
                            acc[rt][ct], false, false);
            }
        }
    }
    __syncthreads();
    #pragma unroll
    for (int ct = 0; ct < 2; ++ct) {
        float inv = 1.0f / s_ls[ct * 16 + l16];
        int n = n0 + ct * 16 + l16;
        #pragma unroll
        for (int rt = 0; rt < 4; ++rt) {
            int cbase = wave * 64 + rt * 16 + hi * 8;
            #pragma unroll
            for (int r = 0; r < 8; ++r)
                out[((size_t)(b * OC + cbase + r)) * N + n] = acc[rt][ct][r] * inv;
        }
    }
}

// qv passthrough into the concat half: out[:, 512:1024, :, :] = qv
__global__ __launch_bounds__(256)
void memreader_qv_copy(const f4* __restrict__ qv, f4* __restrict__ out)
{
    size_t i   = (size_t)blockIdx.x * blockDim.x + threadIdx.x;  // 4,194,304 total
    size_t per = (size_t)512 * 4096 / 4;
    size_t bb  = i / per;
    size_t r   = i - bb * per;
    out[bb * ((size_t)1024 * 4096 / 4) + per + r] = qv[i];
}

extern "C" void kernel_launch(void* const* d_in, const int* in_sizes, int n_in,
                              void* d_out, int out_size, void* d_ws, size_t ws_size,
                              hipStream_t stream) {
    (void)in_sizes; (void)n_in; (void)out_size;
    const float* mk = (const float*)d_in[0];
    const float* qk = (const float*)d_in[1];
    const float* mv = (const float*)d_in[2];
    const float* qv = (const float*)d_in[3];
    float* out = (float*)d_out;

    // workspace layout (bytes)
    const size_t off_mvb = 0;                       // 33,554,432 (mv bf16)
    const size_t off_mkt = 33554432;                //  4,194,304 (mk^T bf16)
    const size_t off_qkt = 37748736;                //  4,194,304 (qk^T bf16)
    const size_t off_asq = 41943040;                //    131,072 (|mk|^2 f32)
    const size_t need    = 42074112;

    if (d_ws && ws_size >= need) {
        char* ws = (char*)d_ws;
        kprep_mv_bf16<<<16384, 256, 0, stream>>>((const f4*)mv, (u2*)(ws + off_mvb));
        kprep_transpose<<<dim3(64, 1, 8), 256, 0, stream>>>(
            mk, (unsigned short*)(ws + off_mkt), (float*)(ws + off_asq));
        kprep_transpose<<<dim3(64, 1, 8), 256, 0, stream>>>(
            qk, (unsigned short*)(ws + off_qkt), (float*)nullptr);
        memreader_flash_fast<<<dim3(4096 / 64, 1, 8), 512, 0, stream>>>(
            (const unsigned short*)(ws + off_mkt),
            (const unsigned short*)(ws + off_qkt),
            (const unsigned short*)(ws + off_mvb),
            (const float*)(ws + off_asq), out);
    } else {
        memreader_flash_fallback<<<dim3(4096 / 32, 1, 8), 256, 0, stream>>>(
            mk, qk, mv, out);
    }

    memreader_qv_copy<<<(8u * 512u * 4096u / 4u) / 256u, 256, 0, stream>>>(
        (const f4*)qv, (f4*)out);
}
